// Attention_62388694942101
// MI455X (gfx1250) — compile-verified, matrix operand order checked by reading, not run on hardware
//
#include <hip/hip_runtime.h>
#include <math.h>

// ---------------------------------------------------------------------------
// MLA attention prefill for MI455X (gfx1250), bf16 WMMA everywhere.
// B=2, T=2048, C=2048, C_latent=512, H=16, dh=128.  All GEMM dims are
// multiples of 64 so tiles need no bounds checks.
// ---------------------------------------------------------------------------

typedef __bf16 bf16_t;
typedef bf16_t bf16x8  __attribute__((ext_vector_type(8)));
typedef bf16_t v16bf   __attribute__((ext_vector_type(16)));
typedef float  v8f     __attribute__((ext_vector_type(8)));

#define CONST_B   2
#define CONST_T   2048
#define CONST_C   2048
#define CONST_CL  512
#define CONST_H   16
#define CONST_DH  128
#define CONST_M   (CONST_B * CONST_T)   // 4096 rows in all GEMMs

// ---- fragment helpers ------------------------------------------------------
// 16-bit A/B operand layout (ISA 7.12.2): lane<16 holds K chunks {c..c+7} and
// {c+16..c+23}; lane>=16 holds {c+8..c+15} and {c+24..c+31}.  Caller passes a
// pointer to the first 8-element chunk; second chunk is +16 elements.
__device__ inline v16bf load_ab_frag(const bf16_t* p) {
  bf16x8 lo = *(const bf16x8*)(p);
  bf16x8 hi = *(const bf16x8*)(p + 16);
  v16bf r;
#pragma unroll
  for (int i = 0; i < 8; ++i) { r[i] = lo[i]; r[i + 8] = hi[i]; }
  return r;
}

__device__ inline v8f wmma_bf16(v16bf a, v16bf b, v8f c) {
  return __builtin_amdgcn_wmma_f32_16x16x32_bf16(
      /*neg_a=*/false, a, /*neg_b=*/false, b,
      /*c_mod=*/(short)0, c, /*reuse_a=*/false, /*reuse_b=*/false);
}

// 16-lane reductions (stay inside each half of the wave32: xor masks < 16).
__device__ inline float hmax16(float v) {
#pragma unroll
  for (int m = 1; m < 16; m <<= 1) v = fmaxf(v, __shfl_xor(v, m, 32));
  return v;
}
__device__ inline float hsum16(float v) {
#pragma unroll
  for (int m = 1; m < 16; m <<= 1) v += __shfl_xor(v, m, 32);
  return v;
}

// ---- fp32 -> bf16 cast -----------------------------------------------------
__global__ void cvt_f32_bf16(const float* __restrict__ in,
                             bf16_t* __restrict__ out, int n) {
  int i = blockIdx.x * blockDim.x + threadIdx.x;
  if (i < n) out[i] = (bf16_t)in[i];
}

// ---- generic WMMA GEMM: out[M,N] = A[M,K] @ W[N,K]^T (+bias) ---------------
// Block = 128 threads = 4 waves; wave tile 32x32 (2x2 accumulators); block
// tile 64x64.  Optional fp32 / bf16 / bf16-transposed outputs.
__global__ void __launch_bounds__(128)
gemm_wmma_bf16(const bf16_t* __restrict__ A, const bf16_t* __restrict__ W,
               const float* __restrict__ bias,
               float* __restrict__ outF, bf16_t* __restrict__ outB,
               bf16_t* __restrict__ outBT, int M, int N, int K) {
  const int lane = threadIdx.x & 31;
  const int wave = threadIdx.x >> 5;
  const int lh   = lane >> 4;     // which half of the wave
  const int l16  = lane & 15;
  const int m0 = blockIdx.y * 64 + (wave >> 1) * 32;
  const int n0 = blockIdx.x * 64 + (wave & 1) * 32;

  v8f zero = {};
  v8f acc[2][2];
  acc[0][0] = zero; acc[0][1] = zero; acc[1][0] = zero; acc[1][1] = zero;

  for (int k0 = 0; k0 < K; k0 += 32) {
    const bf16_t* pa = A + (size_t)(m0 + l16) * K + k0 + lh * 8;
    const bf16_t* pb = W + (size_t)(n0 + l16) * K + k0 + lh * 8;
    __builtin_prefetch(pa + 32, 0, 3);          // global_prefetch_b8
    __builtin_prefetch(pb + 32, 0, 3);
    v16bf a0 = load_ab_frag(pa);
    v16bf a1 = load_ab_frag(pa + (size_t)16 * K);
    v16bf b0 = load_ab_frag(pb);
    v16bf b1 = load_ab_frag(pb + (size_t)16 * K);
    acc[0][0] = wmma_bf16(a0, b0, acc[0][0]);
    acc[0][1] = wmma_bf16(a0, b1, acc[0][1]);
    acc[1][0] = wmma_bf16(a1, b0, acc[1][0]);
    acc[1][1] = wmma_bf16(a1, b1, acc[1][1]);
  }

  float bn[2];
  bn[0] = bias ? bias[n0 + l16]      : 0.0f;
  bn[1] = bias ? bias[n0 + 16 + l16] : 0.0f;

  // C/D layout: VGPR v -> row m0t + v + 8*lh, column n0t + l16.
#pragma unroll
  for (int mt = 0; mt < 2; ++mt) {
#pragma unroll
    for (int nt = 0; nt < 2; ++nt) {
#pragma unroll
      for (int v = 0; v < 8; ++v) {
        int m = m0 + mt * 16 + lh * 8 + v;
        int n = n0 + nt * 16 + l16;
        float val = acc[mt][nt][v] + bn[nt];
        if (outF)  outF[(size_t)m * N + n]  = val;
        if (outB)  outB[(size_t)m * N + n]  = (bf16_t)val;
        if (outBT) outBT[(size_t)n * M + m] = (bf16_t)val;  // transposed
      }
    }
  }
}

// ---- causal flash attention ------------------------------------------------
// Grid (T/64, H, B); block = 128 threads = 4 waves.  Wave w owns 16 query
// rows q0 = 64*blockIdx.x + 16*w and streams keys 32 at a time up to its own
// causal limit (per-wave trip counts -> no block barriers; per-wave LDS).
// Q,K are [B*T, C] bf16; Vt is [C, B*T] bf16 (pre-transposed); O is [B*T, C].
__global__ void __launch_bounds__(128)
mla_flash_attn(const bf16_t* __restrict__ Q, const bf16_t* __restrict__ Km,
               const bf16_t* __restrict__ Vt, bf16_t* __restrict__ O) {
  __shared__ bf16_t plds[4][16][32];   // per-wave P tile (16 q x 32 k)

  const int lane = threadIdx.x & 31;
  const int wave = threadIdx.x >> 5;
  const int lh   = lane >> 4;
  const int l16  = lane & 15;
  const int b = blockIdx.z;
  const int h = blockIdx.y;
  const int q0 = blockIdx.x * 64 + wave * 16;

  const float scale = 0.08838834764831845f;    // 1/sqrt(128)

  // Q fragments for all of dh=128 (4 k-steps of 32), kept in registers.
  v16bf qf[4];
  const size_t qbase = ((size_t)b * CONST_T + q0 + l16) * CONST_C + h * CONST_DH;
#pragma unroll
  for (int ks = 0; ks < 4; ++ks)
    qf[ks] = load_ab_frag(Q + qbase + ks * 32 + lh * 8);

  v8f zero = {};
  v8f o[8];
#pragma unroll
  for (int t = 0; t < 8; ++t) o[t] = zero;
  float mi[8], li[8];
#pragma unroll
  for (int v = 0; v < 8; ++v) { mi[v] = -1e30f; li[v] = 0.0f; }

  const int ktiles = (q0 + 15) / 32 + 1;       // keys 0 .. q0+15 inclusive
  for (int kt = 0; kt < ktiles; ++kt) {
    const int kb = kt * 32;

    // ---- S = Q @ K^T for a 16x32 score strip (two 16x16 C tiles) ----
    v8f s0 = zero, s1 = zero;
    const bf16_t* kbase =
        Km + ((size_t)b * CONST_T + kb) * CONST_C + h * CONST_DH;
#pragma unroll
    for (int ks = 0; ks < 4; ++ks) {
      v16bf k0f = load_ab_frag(kbase + (size_t)l16 * CONST_C + ks * 32 + lh * 8);
      v16bf k1f =
          load_ab_frag(kbase + (size_t)(16 + l16) * CONST_C + ks * 32 + lh * 8);
      s0 = wmma_bf16(qf[ks], k0f, s0);
      s1 = wmma_bf16(qf[ks], k1f, s1);
    }

    const bool need_mask = (kb + 31 > q0);
    // ---- online softmax update, per owned row (row = lh*8 + v) ----
#pragma unroll
    for (int v = 0; v < 8; ++v) {
      float a = s0[v] * scale;
      float c = s1[v] * scale;
      if (need_mask) {
        int qrow = q0 + lh * 8 + v;
        if (kb + l16 > qrow)      a = -1e30f;
        if (kb + 16 + l16 > qrow) c = -1e30f;
      }
      float rm = hmax16(fmaxf(a, c));
      float mn = fmaxf(mi[v], rm);
      float alpha = __expf(mi[v] - mn);
      float p0 = __expf(a - mn);
      float p1 = __expf(c - mn);
      float rs = hsum16(p0 + p1);
      li[v] = li[v] * alpha + rs;
      mi[v] = mn;
#pragma unroll
      for (int t = 0; t < 8; ++t) o[t][v] = o[t][v] * alpha;
      // stash P in LDS so it can be re-read in A-fragment layout
      plds[wave][lh * 8 + v][l16]      = (bf16_t)p0;
      plds[wave][lh * 8 + v][16 + l16] = (bf16_t)p1;
    }

    // DS ops are in-order within a wave; wait + compiler fence before reread.
    asm volatile("s_wait_dscnt 0" ::: "memory");

    v16bf pf = load_ab_frag(&plds[wave][l16][lh * 8]);

    // ---- O += P @ V  (V^T rows are contiguous in key index) ----
    const bf16_t* vbase =
        Vt + (size_t)h * CONST_DH * CONST_M + (size_t)b * CONST_T + kb;
#pragma unroll
    for (int t = 0; t < 8; ++t) {
      v16bf vf = load_ab_frag(vbase + (size_t)(t * 16 + l16) * CONST_M + lh * 8);
      o[t] = wmma_bf16(pf, vf, o[t]);
    }
  }

  // ---- epilogue: divide by row sums, write merged-head output ----
#pragma unroll
  for (int v = 0; v < 8; ++v) {
    float inv = 1.0f / li[v];
    size_t row = ((size_t)b * CONST_T + q0 + lh * 8 + v) * CONST_C + h * CONST_DH;
#pragma unroll
    for (int t = 0; t < 8; ++t)
      O[row + t * 16 + l16] = (bf16_t)(o[t][v] * inv);
  }
}

// ---------------------------------------------------------------------------
extern "C" void kernel_launch(void* const* d_in, const int* in_sizes, int n_in,
                              void* d_out, int out_size, void* d_ws,
                              size_t ws_size, hipStream_t stream) {
  const float* X     = (const float*)d_in[0];
  const float* WQ_w  = (const float*)d_in[1];
  const float* WQ_b  = (const float*)d_in[2];
  const float* WO_w  = (const float*)d_in[3];
  const float* WO_b  = (const float*)d_in[4];
  const float* WKc_w = (const float*)d_in[5];
  const float* WKu_w = (const float*)d_in[6];
  const float* WVc_w = (const float*)d_in[7];
  const float* WVu_w = (const float*)d_in[8];
  float* out = (float*)d_out;

  const int M = CONST_M, C = CONST_C, CL = CONST_CL;

  // carve bf16 workspace (~118 MB total; every chunk 16B-aligned)
  char* w = (char*)d_ws;
  auto carve = [&](size_t elems) {
    bf16_t* p = (bf16_t*)w;
    w += elems * sizeof(bf16_t);
    return p;
  };
  bf16_t* Xb   = carve((size_t)M * C);
  bf16_t* WQb  = carve((size_t)C * C);
  bf16_t* WOb  = carve((size_t)C * C);
  bf16_t* WKcb = carve((size_t)CL * C);
  bf16_t* WKub = carve((size_t)C * CL);
  bf16_t* WVcb = carve((size_t)CL * C);
  bf16_t* WVub = carve((size_t)C * CL);
  bf16_t* Qb   = carve((size_t)M * C);
  bf16_t* Kb   = carve((size_t)M * C);
  bf16_t* Vtb  = carve((size_t)C * M);   // transposed V: [C, B*T]
  bf16_t* KLb  = carve((size_t)M * CL);
  bf16_t* VLb  = carve((size_t)M * CL);
  bf16_t* Ob   = carve((size_t)M * C);

  // d_out layout: out [M*C], K_latent [M*CL], V_latent [M*CL]
  float* out_main = out;
  float* out_KL   = out + (size_t)M * C;
  float* out_VL   = out + (size_t)M * C + (size_t)M * CL;

  auto cvt = [&](const float* src, bf16_t* dst, int n) {
    cvt_f32_bf16<<<(n + 255) / 256, 256, 0, stream>>>(src, dst, n);
  };
  cvt(X, Xb, M * C);
  cvt(WQ_w, WQb, C * C);
  cvt(WO_w, WOb, C * C);
  cvt(WKc_w, WKcb, CL * C);
  cvt(WKu_w, WKub, C * CL);
  cvt(WVc_w, WVcb, CL * C);
  cvt(WVu_w, WVub, C * CL);

  dim3 blk(128);
  auto grid = [&](int N_) { return dim3(N_ / 64, M / 64); };

  // K_latent = X @ WKc^T  (fp32 to d_out, bf16 to ws)
  gemm_wmma_bf16<<<grid(CL), blk, 0, stream>>>(Xb, WKcb, nullptr, out_KL, KLb,
                                               nullptr, M, CL, C);
  // V_latent = X @ WVc^T
  gemm_wmma_bf16<<<grid(CL), blk, 0, stream>>>(Xb, WVcb, nullptr, out_VL, VLb,
                                               nullptr, M, CL, C);
  // Q = X @ WQ^T + b
  gemm_wmma_bf16<<<grid(C), blk, 0, stream>>>(Xb, WQb, WQ_b, nullptr, Qb,
                                              nullptr, M, C, C);
  // K = K_latent @ WKu^T
  gemm_wmma_bf16<<<grid(C), blk, 0, stream>>>(KLb, WKub, nullptr, nullptr, Kb,
                                              nullptr, M, C, CL);
  // V = V_latent @ WVu^T  (stored transposed for the flash kernel)
  gemm_wmma_bf16<<<grid(C), blk, 0, stream>>>(VLb, WVub, nullptr, nullptr,
                                              nullptr, Vtb, M, C, CL);
  // causal flash attention, merged-head output
  mla_flash_attn<<<dim3(CONST_T / 64, CONST_H, CONST_B), blk, 0, stream>>>(
      Qb, Kb, Vtb, Ob);
  // out = attn_out @ WO^T + b  (fp32 to d_out)
  gemm_wmma_bf16<<<grid(C), blk, 0, stream>>>(Ob, WOb, WO_b, out_main, nullptr,
                                              nullptr, M, C, C);
}
